// NystromNonLinearKernel_86766929314328
// MI455X (gfx1250) — matrix operand order repaired and checked by reading, not compiled, not run on hardware
//
#include <hip/hip_runtime.h>
#include <hip/hip_bf16.h>
#include <math.h>

#define B_    8
#define CIN_  16
#define HID_  128
#define S_    32
#define H_    256
#define W_    256
#define RS_   (S_*S_)   // 1024

typedef __attribute__((ext_vector_type(16))) _Float16 v16h;
typedef __attribute__((ext_vector_type(8)))  _Float16 v8h;
typedef __attribute__((ext_vector_type(8)))  float    v8f;
typedef __attribute__((ext_vector_type(4)))  float    v4f;

static __device__ __forceinline__ v16h cat8(v8h lo, v8h hi){
  v16h r;
#pragma unroll
  for (int i=0;i<8;++i){ r[i]=lo[i]; r[i+8]=hi[i]; }
  return r;
}

// CDNA5 async global->LDS copy (16B), tracked by ASYNCcnt (ISA 08_async_tensor §4).
// VDST = LDS byte offset VGPR (generic LDS addr truncates to DS offset, ISA §10.2).
static __device__ __forceinline__ void async_cp16(const void* g, void* l){
  unsigned           ldsOff = (unsigned)(size_t)l;
  unsigned long long ga     = (unsigned long long)(size_t)g;
  asm volatile("global_load_async_to_lds_b128 %0, %1, off"
               :: "v"(ldsOff), "v"(ga) : "memory");
}

// Branchless tanh: native v_tanh_f32 on gfx1250 when exposed; else exp-based (v_exp_f32).
static __device__ __forceinline__ float tanh_fast(float z){
#if __has_builtin(__builtin_amdgcn_tanhf)
  return __builtin_amdgcn_tanhf(z);
#elif __has_builtin(__builtin_amdgcn_tanh_f32)
  return __builtin_amdgcn_tanh_f32(z);
#else
  float zc = fminf(fmaxf(z, -9.f), 9.f);   // avoid inf/inf
  float e  = __expf(2.f*zc);
  return (e - 1.f) / (e + 1.f);
#endif
}
// gelu(x) ~= 0.5x(1+tanh(0.79788456(x+0.044715x^3))); error << f16 quantization of hidden.
static __device__ __forceinline__ float gelu_fast(float x){
  float x3 = x*x*x;
  float z  = fmaf(0.035677408136f, x3, 0.7978845608028654f*x);
  return 0.5f*x*(1.f + tanh_fast(z));
}

// ---------------- K0: pre-sum W2 over cout (m), transpose to [n][h] f16; b2s[n] ----
__global__ void prep_kernel(const float* __restrict__ W2, const float* __restrict__ b2,
                            _Float16* __restrict__ w2sT, float* __restrict__ b2s){
  int t = threadIdx.x;
  for (int e = t; e < CIN_*HID_; e += blockDim.x){
    int n = e >> 7;          // 0..15
    int h = e & 127;         // 0..127
    float s = 0.f;
#pragma unroll
    for (int m=0;m<16;++m) s += W2[h*256 + m*16 + n];
    w2sT[n*HID_ + h] = (_Float16)s;
  }
  if (t < CIN_){
    float s = 0.f;
#pragma unroll
    for (int m=0;m<16;++m) s += b2[m*16 + t];
    b2s[t] = s;
  }
}

// ---------------- K1: bilinear grid-sample, pack to WMMA-B-friendly f16 [n][b16][rs] ----
__global__ void sample_kernel(const float* __restrict__ v, const float* __restrict__ grid,
                              _Float16* __restrict__ sampB){
  int idx = blockIdx.x*blockDim.x + threadIdx.x;   // 16*16*1024
  int b16 = idx >> 14;
  int n   = (idx >> 10) & 15;
  int rs  = idx & 1023;
  float val = 0.f;
  if (b16 < B_){
    float gx = grid[rs*2+0], gy = grid[rs*2+1];
    float x = (gx+1.f)*(W_*0.5f) - 0.5f;
    float y = (gy+1.f)*(H_*0.5f) - 0.5f;
    float x0f = floorf(x), y0f = floorf(y);
    int   x0 = (int)x0f,  y0 = (int)y0f;
    float wx = x - x0f,   wy = y - y0f;
    const float* vp = v + ((size_t)(b16*CIN_ + n))*(size_t)(H_*W_);
    auto tap = [&](int xi,int yi)->float{
      if (xi < 0 || xi > W_-1 || yi < 0 || yi > H_-1) return 0.f;
      return vp[yi*W_ + xi];
    };
    val = tap(x0  ,y0  )*(1.f-wx)*(1.f-wy)
        + tap(x0+1,y0  )*wx      *(1.f-wy)
        + tap(x0  ,y0+1)*(1.f-wx)*wy
        + tap(x0+1,y0+1)*wx      *wy;
  }
  sampB[((size_t)(n*16 + b16))*RS_ + rs] = (_Float16)val;   // b16>=8 lanes write zero padding
}

// ---------------- K1b: deterministic per-(b,n) sum of sample over rs ----
__global__ void ssum_kernel(const _Float16* __restrict__ sampB, float* __restrict__ Ssum){
  int bn = threadIdx.x;
  if (bn >= B_*CIN_) return;
  int b = bn >> 4, n = bn & 15;
  const _Float16* p = sampB + ((size_t)(n*16 + b))*RS_;
  float s = 0.f;
  for (int i=0;i<RS_;++i) s += (float)p[i];
  Ssum[bn] = s;
}

// ---------------- K2: fused gelu-MLP + double WMMA contraction ----
// grid: 64 blocks (16 pq each), 256 threads = 8 wave32.
__global__ __launch_bounds__(256) void fused_kernel(
    const float* __restrict__ grid, const float* __restrict__ W1, const float* __restrict__ b1,
    const _Float16* __restrict__ w2sT, const float* __restrict__ b2s,
    const float* __restrict__ Ssum, const _Float16* __restrict__ sampB,
    float* __restrict__ outS){
  extern __shared__ char smem[];
  _Float16* sh_hidden = (_Float16*)smem;                               // 512*128 f16 = 128KB
  _Float16* sh_ksum   = (_Float16*)(smem + 131072);                    // [n=16][pq=16][rs=32] = 16KB
  _Float16* sh_samp   = (_Float16*)(smem + 147456);                    // 2 x [row=256][rs=32] = 32KB
  float*    sh_grid   = (float*)(smem + 180224);                       // 2048 f32 = 8KB
  float*    sh_w1     = sh_grid + 2048;                                // [w1x|w1y|b1] = 384 f32

  const int tid  = threadIdx.x;
  const int lane = tid & 31;
  const int w    = tid >> 5;         // wave id 0..7
  const int m15  = lane & 15;
  const int hiH  = lane >> 4;        // 0/1: lane half selects K sub-block
  const int pq0  = blockIdx.x * 16;

  for (int i = tid; i < 2048; i += 256) sh_grid[i] = grid[i];
  for (int i = tid; i < 384;  i += 256) sh_w1[i]  = (i < 256) ? W1[i] : b1[i-256];

  // Prologue: async-stage rs-chunk 0 of the B operand into buffer 0 (row = tid, 64B/row).
  {
    _Float16*        dst = sh_samp + (size_t)tid*32;
    const _Float16*  src = sampB  + (size_t)tid*RS_;
#pragma unroll
    for (int q=0;q<4;++q) async_cp16(src + q*8, dst + q*8);
  }
  __syncthreads();

  // Hoist WMMA#1 B operand (W2sT) for all 4 K-chunks: B[K=h][N=n], lane=N.
  v16h Bw[4];
#pragma unroll
  for (int kc=0; kc<4; ++kc){
    const _Float16* p = w2sT + m15*HID_ + kc*32 + hiH*8;
    Bw[kc] = cat8(*(const v8h*)p, *(const v8h*)(p+16));
  }

  v8f acc0 = {}, acc1 = {};

  for (int t=0; t<32; ++t){
    const int rs0 = t*32;

    // ---- Phase A: hidden tile [row = rs_local*16 + pq_local][h] = gelu(diff@W1+b1), f16
    // Two rows per thread; h blocked by 8 so weights load as 128-bit LDS ops shared by
    // both rows, and each row's 8 halfs store as one ds_store_b128. Branchless gelu.
    float dxr[2], dyr[2];
#pragma unroll
    for (int rr=0; rr<2; ++rr){
      int row = tid + rr*256;
      int rsl = row >> 4, pql = row & 15;
      int rsg = rs0 + rsl, pqg = pq0 + pql;
      dxr[rr] = sh_grid[rsg*2+0] - sh_grid[pqg*2+0];
      dyr[rr] = sh_grid[rsg*2+1] - sh_grid[pqg*2+1];
    }
    for (int hb=0; hb<HID_; hb+=8){
      v4f wxA = *(const v4f*)&sh_w1[hb],       wxB = *(const v4f*)&sh_w1[hb+4];
      v4f wyA = *(const v4f*)&sh_w1[128+hb],   wyB = *(const v4f*)&sh_w1[128+hb+4];
      v4f bbA = *(const v4f*)&sh_w1[256+hb],   bbB = *(const v4f*)&sh_w1[256+hb+4];
#pragma unroll
      for (int rr=0; rr<2; ++rr){
        float dx = dxr[rr], dy = dyr[rr];
        v8h o;
#pragma unroll
        for (int k=0; k<8; ++k){
          float wx = (k<4) ? wxA[k&3] : wxB[k&3];
          float wy = (k<4) ? wyA[k&3] : wyB[k&3];
          float bb = (k<4) ? bbA[k&3] : bbB[k&3];
          float pre = fmaf(dx, wx, fmaf(dy, wy, bb));
          o[k] = (_Float16)gelu_fast(pre);
        }
        *(v8h*)(sh_hidden + (size_t)(tid + rr*256)*HID_ + hb) = o;
      }
    }
    __syncthreads();

    // Kick off async staging of the NEXT rs-chunk into the other buffer, overlapping
    // with this iteration's WMMA work (ASYNCcnt-tracked, waited below).
    if (t < 31){
      _Float16*        dst = sh_samp + ((t+1)&1)*8192 + (size_t)tid*32;
      const _Float16*  src = sampB  + (size_t)tid*RS_ + (rs0 + 32);
#pragma unroll
      for (int q=0;q<4;++q) async_cp16(src + q*8, dst + q*8);
    }

    // ---- Phase B: ksum = hidden @ W2sT  (32 row-tiles, 4 per wave; K=128 -> 4 WMMAs each)
#pragma unroll
    for (int i=0;i<4;++i){
      int rt  = w*4 + i;              // row tile == rs_local (rows ordered rs_local*16+pq)
      int row = rt*16 + m15;          // A: lane -> M (pq_local), hiH picks K sub-block
      // Preload all 4 A fragments so the ds_load_b128s issue back-to-back.
      const _Float16* ap = sh_hidden + row*HID_ + hiH*8;
      v16h a0 = cat8(*(const v8h*)(ap +  0), *(const v8h*)(ap + 16));
      v16h a1 = cat8(*(const v8h*)(ap + 32), *(const v8h*)(ap + 48));
      v16h a2 = cat8(*(const v8h*)(ap + 64), *(const v8h*)(ap + 80));
      v16h a3 = cat8(*(const v8h*)(ap + 96), *(const v8h*)(ap +112));
      v8f c = {};
      c = __builtin_amdgcn_wmma_f32_16x16x32_f16(false, a0, false, Bw[0], (short)0, c, false, false);
      c = __builtin_amdgcn_wmma_f32_16x16x32_f16(false, a1, false, Bw[1], (short)0, c, false, false);
      c = __builtin_amdgcn_wmma_f32_16x16x32_f16(false, a2, false, Bw[2], (short)0, c, false, false);
      c = __builtin_amdgcn_wmma_f32_16x16x32_f16(false, a3, false, Bw[3], (short)0, c, false, false);
      // D: lane col N = n = m15, row M = j + 8*hiH = pq_local. Transpose into [n][pq][rs].
      int n = m15;
#pragma unroll
      for (int j=0;j<8;++j)
        sh_ksum[(n*16 + (j + 8*hiH))*32 + rt] = (_Float16)c[j];
    }

    // Current B tile must have landed in LDS: allow only the 4 just-issued (next-tile)
    // async ops to remain outstanding (async loads complete in order).
    if (t < 31) asm volatile("s_wait_asynccnt 4" ::: "memory");
    else        asm volatile("s_wait_asynccnt 0" ::: "memory");
    __syncthreads();

    // ---- Phase C: per-channel contraction out[pq,b] += ksum[pq,rs] @ sample[rs,b]
    const _Float16* bufc = sh_samp + (t&1)*8192;
#pragma unroll
    for (int nl=0;nl<2;++nl){
      int n = w*2 + nl;
      const _Float16* ap = sh_ksum + (n*16 + m15)*32 + hiH*8;          // A: M=pq, K=rs32
      v16h a = cat8(*(const v8h*)ap, *(const v8h*)(ap+16));
      const _Float16* bp = bufc + (n*16 + m15)*32 + hiH*8;             // B: N=b16, from LDS
      v16h bm = cat8(*(const v8h*)bp, *(const v8h*)(bp+16));
      if (nl==0) acc0 = __builtin_amdgcn_wmma_f32_16x16x32_f16(false,a,false,bm,(short)0,acc0,false,false);
      else       acc1 = __builtin_amdgcn_wmma_f32_16x16x32_f16(false,a,false,bm,(short)0,acc1,false,false);
    }
    // Next-iter Phase A writes sh_hidden (disjoint); its barrier orders Phase C reads
    // of sh_ksum/sh_samp vs next Phase B/async writes.
  }

  // ---- Writeback: out[b,n,pq] = (acc + b2s[n]*Ssum[b,n]) / S
  const float invS = 1.0f/32.0f;
#pragma unroll
  for (int nl=0;nl<2;++nl){
    int n = w*2 + nl;
    v8f d = nl ? acc1 : acc0;
    int b = m15;                       // D col N = b (only b<8 valid; 8..15 were zero-padded)
    if (b < B_){
      float add = b2s[n]*Ssum[b*16+n];
#pragma unroll
      for (int j=0;j<8;++j){
        int pqg = pq0 + j + 8*hiH;
        outS[((size_t)(b*16+n))*RS_ + pqg] = (d[j] + add)*invS;
      }
    }
  }
}

// ---------------- K3: half-pixel bilinear 32x32 -> 256x256 (align_corners=False) ----
__global__ void resize_kernel(const float* __restrict__ outS, float* __restrict__ u){
  int idx = blockIdx.x*blockDim.x + threadIdx.x;   // 8*16*256*256
  int x  = idx & 255;
  int y  = (idx >> 8) & 255;
  int bn = idx >> 16;
  float sx = (x + 0.5f)*0.125f - 0.5f;
  float sy = (y + 0.5f)*0.125f - 0.5f;
  float x0f = floorf(sx), y0f = floorf(sy);
  float wx = sx - x0f, wy = sy - y0f;
  int ix0 = (int)x0f;     ix0 = ix0 < 0 ? 0 : (ix0 > 31 ? 31 : ix0);
  int ix1 = (int)x0f + 1; ix1 = ix1 < 0 ? 0 : (ix1 > 31 ? 31 : ix1);
  int iy0 = (int)y0f;     iy0 = iy0 < 0 ? 0 : (iy0 > 31 ? 31 : iy0);
  int iy1 = (int)y0f + 1; iy1 = iy1 < 0 ? 0 : (iy1 > 31 ? 31 : iy1);
  const float* p = outS + (size_t)bn*RS_;
  float v00 = p[iy0*32+ix0], v10 = p[iy0*32+ix1];
  float v01 = p[iy1*32+ix0], v11 = p[iy1*32+ix1];
  u[idx] = v00*(1.f-wx)*(1.f-wy) + v10*wx*(1.f-wy) + v01*(1.f-wx)*wy + v11*wx*wy;
}

// ---------------- launch ----------------
extern "C" void kernel_launch(void* const* d_in, const int* in_sizes, int n_in,
                              void* d_out, int out_size, void* d_ws, size_t ws_size,
                              hipStream_t stream) {
  (void)in_sizes; (void)n_in; (void)out_size; (void)ws_size;
  const float* v    = (const float*)d_in[0];
  const float* grid = (const float*)d_in[1];
  const float* W1   = (const float*)d_in[2];
  const float* b1   = (const float*)d_in[3];
  const float* W2   = (const float*)d_in[4];
  const float* b2   = (const float*)d_in[5];

  char* ws = (char*)d_ws;
  _Float16* sampB = (_Float16*)(ws + 0);          // 16*16*1024 f16 = 512KB
  _Float16* w2sT  = (_Float16*)(ws + 524288);     // 16*128 f16
  float*    b2s   = (float*)(ws + 528384);        // 16 f32
  float*    Ssum  = (float*)(ws + 528448);        // 128 f32
  float*    outS  = (float*)(ws + 528960);        // 8*16*1024 f32 = 512KB
  float*    u     = (float*)d_out;

  prep_kernel  <<<1,    256, 0, stream>>>(W2, b2, w2sT, b2s);
  sample_kernel<<<1024, 256, 0, stream>>>(v, grid, sampB);
  ssum_kernel  <<<1,    128, 0, stream>>>(sampB, Ssum);

  // LDS: 128K hidden + 16K ksum + 32K double-buffered B tile + 8K grid + 1.5K W1 = ~185.5KB
  size_t shmem = 131072 + 16384 + 32768 + 8192 + 1536;
  fused_kernel <<<64,   256, shmem, stream>>>(grid, W1, b1, w2sT, b2s, Ssum, sampB, outS);

  resize_kernel<<<32768,256, 0, stream>>>(outS, u);
}
